// MultiHeadAttention_46961172414517
// MI455X (gfx1250) — matrix operand order repaired and checked by reading, not compiled, not run on hardware
//
#include <hip/hip_runtime.h>

// ---------------------------------------------------------------------------
// MI455X (gfx1250) multi-head attention + rel-pos bias + softmax + LayerNorm.
// All GEMM-like math uses v_wmma_f32_16x16x32_bf16 (wave32 WMMA), softmax/LN
// stay in f32. probs (512MB) is written exactly once; PV consumes the probs
// tile through an LDS bounce (no HBM re-read).
// ---------------------------------------------------------------------------

typedef __attribute__((ext_vector_type(16))) __bf16 v16bf;
typedef __attribute__((ext_vector_type(8)))  __bf16 v8bf;
typedef __attribute__((ext_vector_type(8)))  float  v8f;

#define WMMA_BF16(A, B, C) \
  __builtin_amdgcn_wmma_f32_16x16x32_bf16(false, (A), false, (B), (short)0, (C), false, false)

static __device__ __forceinline__ v16bf join16(v8bf lo, v8bf hi) {
  return __builtin_shufflevector(lo, hi, 0,1,2,3,4,5,6,7,8,9,10,11,12,13,14,15);
}

// --------------------------- f32 -> bf16 convert ---------------------------
__global__ __launch_bounds__(256) void cvt_f32_bf16(const float* __restrict__ in,
                                                    __bf16* __restrict__ out, int n) {
  int i = blockIdx.x * 256 + threadIdx.x;
  if (i < n) out[i] = (__bf16)in[i];
}

// ------------------------------- GEMM --------------------------------------
// Y[row, col] = sum_k X[row,k] * W[col,k] + bias[col]   (torch Linear x@W.T+b)
// block = 256 thr = 8 waves; each wave computes a 32x32 tile (2x2 WMMA tiles),
// block tile = 64(M) x 128(N). M=4096, N=1024, K=1024.
// mode: 0 -> bf16 out as (b,h,s,d);  2 -> bf16 out transposed (b,h,d,s);
//       3 -> f32 out row-major (attn_out).
__global__ __launch_bounds__(256) void gemm_bf16(const __bf16* __restrict__ X,
                                                 const __bf16* __restrict__ W,
                                                 const float*  __restrict__ bias,
                                                 __bf16* __restrict__ outb,
                                                 float*  __restrict__ outf,
                                                 int mode) {
  const int lane = threadIdx.x & 31;
  const int wave = threadIdx.x >> 5;
  const int half = lane >> 4;
  const int l16  = lane & 15;
  const int mbase = blockIdx.y * 64  + (wave & 1)  * 32;
  const int nbase = blockIdx.x * 128 + (wave >> 1) * 32;

  v8f c[2][2] = {};
  for (int k0 = 0; k0 < 1024; k0 += 32) {
    v16bf A[2], Bm[2];
#pragma unroll
    for (int mi = 0; mi < 2; ++mi) {
      const __bf16* xr = X + (size_t)(mbase + mi * 16 + l16) * 1024 + k0;
      A[mi] = *(const v16bf*)(xr + half * 16);        // K = half*16 + 0..15
    }
#pragma unroll
    for (int ni = 0; ni < 2; ++ni) {
      const __bf16* wr = W + (size_t)(nbase + ni * 16 + l16) * 1024 + k0 + half * 16;
      Bm[ni] = *(const v16bf*)wr;                     // K = half*16 + 0..15
    }
#pragma unroll
    for (int mi = 0; mi < 2; ++mi)
#pragma unroll
      for (int ni = 0; ni < 2; ++ni)
        c[mi][ni] = WMMA_BF16(A[mi], Bm[ni], c[mi][ni]);
  }

#pragma unroll
  for (int mi = 0; mi < 2; ++mi) {
#pragma unroll
    for (int ni = 0; ni < 2; ++ni) {
      const int col = nbase + ni * 16 + l16;
      const float bv = bias[col];
#pragma unroll
      for (int r = 0; r < 8; ++r) {
        const float v = c[mi][ni][r] + bv;
        const int row = mbase + mi * 16 + r + half * 8;
        if (mode == 3) {
          outf[(size_t)row * 1024 + col] = v;
        } else {
          const int bidx = row >> 10, s = row & 1023;
          const int h = col >> 5, d = col & 31;
          size_t idx;
          if (mode == 2) idx = (((size_t)(bidx * 32 + h)) * 32 + d) * 1024 + s;   // V^T
          else           idx = (((size_t)(bidx * 32 + h)) * 1024 + s) * 32 + d;   // (b,h,s,d)
          outb[idx] = (__bf16)v;
        }
      }
    }
  }
}

// ----------------------------- Attention -----------------------------------
// One wave per (b, h, 16-row q tile). Flash-style 3 passes over 64 key tiles:
//   pass1: scores (WMMA, +bias, *scale) -> per-row max (shfl_xor reduce)
//   pass2: recompute -> sum of exp
//   pass3: recompute -> normalized probs -> global write + LDS tile -> PV WMMA
__global__ __launch_bounds__(32) void attn_kernel(const __bf16* __restrict__ Q,
                                                  const __bf16* __restrict__ K,
                                                  const __bf16* __restrict__ VT,
                                                  const float*  __restrict__ dist_emb,
                                                  float* __restrict__ probs,
                                                  __bf16* __restrict__ ctxb) {
  const int lane = threadIdx.x & 31;
  const int half = lane >> 4;
  const int l16  = lane & 15;
  const int qt = blockIdx.x, h = blockIdx.y, b = blockIdx.z;
  const int qb = qt * 16;
  const float scale = 0.17677669529663687f;   // 1/sqrt(32)

  __shared__ float lds_bias[1040];            // rel-pos bias rows [qb, qb+1038]
  __shared__ float ptile[16 * 33];            // 16x32 probs tile, padded stride

  for (int i = threadIdx.x; i < 1039; i += 32)
    lds_bias[i] = dist_emb[(size_t)(qb + i) * 32 + h];
  __syncthreads();                            // single-wave WG: lowers to S_NOP

  // Q tile (A layout), held for the whole kernel.
  const __bf16* qrow = Q + ((((size_t)(b * 32 + h)) * 1024 + qb) + l16) * 32;
  const v16bf Aq = *(const v16bf*)(qrow + half * 16);

  const __bf16* Kbase = K  + ((size_t)(b * 32 + h)) * 1024 * 32;
  const __bf16* Vbase = VT + ((size_t)(b * 32 + h)) * 32 * 1024;
  float* probs_base = probs + (((size_t)(b * 32 + h)) * 1024 + qb) * 1024;

  float rmax[8], rsum[8], rinv[8];
#pragma unroll
  for (int r = 0; r < 8; ++r) { rmax[r] = -3.0e38f; rsum[r] = 0.0f; }

  // ---- pass 1: row max ----
  for (int kt = 0; kt < 64; ++kt) {
    const v16bf Bk = *(const v16bf*)(Kbase + (size_t)(kt * 16 + l16) * 32 + half * 16);
    v8f cc = {};
    cc = WMMA_BF16(Aq, Bk, cc);
    const int kcol = kt * 16 + l16;
#pragma unroll
    for (int r = 0; r < 8; ++r) {
      const int M = r + half * 8;
      const float x = (cc[r] + lds_bias[M + 1023 - kcol]) * scale;
      rmax[r] = fmaxf(rmax[r], x);
    }
  }
#pragma unroll
  for (int r = 0; r < 8; ++r)
    for (int m = 1; m < 16; m <<= 1)
      rmax[r] = fmaxf(rmax[r], __shfl_xor(rmax[r], m, 32));

  // ---- pass 2: row sum of exp ----
  for (int kt = 0; kt < 64; ++kt) {
    const v16bf Bk = *(const v16bf*)(Kbase + (size_t)(kt * 16 + l16) * 32 + half * 16);
    v8f cc = {};
    cc = WMMA_BF16(Aq, Bk, cc);
    const int kcol = kt * 16 + l16;
#pragma unroll
    for (int r = 0; r < 8; ++r) {
      const int M = r + half * 8;
      const float x = (cc[r] + lds_bias[M + 1023 - kcol]) * scale;
      rsum[r] += __expf(x - rmax[r]);
    }
  }
#pragma unroll
  for (int r = 0; r < 8; ++r) {
    for (int m = 1; m < 16; m <<= 1)
      rsum[r] += __shfl_xor(rsum[r], m, 32);
    rinv[r] = 1.0f / rsum[r];
  }

  // ---- pass 3: normalized probs -> global + LDS, then PV WMMA ----
  v8f c0 = {}, c1 = {};
  for (int kt2 = 0; kt2 < 32; ++kt2) {
#pragma unroll
    for (int sub = 0; sub < 2; ++sub) {
      const int kt = kt2 * 2 + sub;
      const v16bf Bk = *(const v16bf*)(Kbase + (size_t)(kt * 16 + l16) * 32 + half * 16);
      v8f cc = {};
      cc = WMMA_BF16(Aq, Bk, cc);
      const int kcol = kt * 16 + l16;
#pragma unroll
      for (int r = 0; r < 8; ++r) {
        const int M = r + half * 8;
        const float x = (cc[r] + lds_bias[M + 1023 - kcol]) * scale;
        const float p = __expf(x - rmax[r]) * rinv[r];
        probs_base[(size_t)M * 1024 + kcol] = p;   // coalesced 64B/half-wave
        ptile[M * 33 + sub * 16 + l16] = p;
      }
    }
    __syncthreads();  // single wave: DS in-order; barrier is S_NOP + dscnt wait

    v16bf Ap;
#pragma unroll
    for (int j = 0; j < 16; ++j) {
      const int kl = half * 16 + j;   // contiguous k order, matches B
      Ap[j] = (__bf16)ptile[l16 * 33 + kl];
    }
    const v16bf B0 = *(const v16bf*)(Vbase + (size_t)(l16)      * 1024 + kt2 * 32 + half * 16);
    const v16bf B1 = *(const v16bf*)(Vbase + (size_t)(16 + l16) * 1024 + kt2 * 32 + half * 16);
    c0 = WMMA_BF16(Ap, B0, c0);
    c1 = WMMA_BF16(Ap, B1, c1);
    __syncthreads();
  }

  // ctx -> (b, s, h*32 + d) bf16 for the O-projection GEMM
#pragma unroll
  for (int r = 0; r < 8; ++r) {
    const int s = qb + r + half * 8;
    const size_t rowoff = ((size_t)b * 1024 + s) * 1024 + h * 32;
    ctxb[rowoff + l16]      = (__bf16)c0[r];
    ctxb[rowoff + 16 + l16] = (__bf16)c1[r];
  }
}

// --------------------------- residual + LayerNorm --------------------------
__global__ __launch_bounds__(256) void ln_kernel(const float* __restrict__ attn_out,
                                                 const float* __restrict__ hs,
                                                 const float* __restrict__ gamma,
                                                 const float* __restrict__ beta,
                                                 float* __restrict__ out) {
  const int row = blockIdx.x, tid = threadIdx.x;
  __shared__ float red[256];
  const float* ar = attn_out + (size_t)row * 1024;
  const float* hr = hs + (size_t)row * 1024;
  float x[4], s = 0.0f;
#pragma unroll
  for (int j = 0; j < 4; ++j) { x[j] = ar[tid + j * 256] + hr[tid + j * 256]; s += x[j]; }
  red[tid] = s; __syncthreads();
  for (int st = 128; st > 0; st >>= 1) { if (tid < st) red[tid] += red[tid + st]; __syncthreads(); }
  const float mu = red[0] * (1.0f / 1024.0f);
  __syncthreads();
  float vs = 0.0f;
#pragma unroll
  for (int j = 0; j < 4; ++j) { const float d = x[j] - mu; vs += d * d; }
  red[tid] = vs; __syncthreads();
  for (int st = 128; st > 0; st >>= 1) { if (tid < st) red[tid] += red[tid + st]; __syncthreads(); }
  const float rstd = rsqrtf(red[0] * (1.0f / 1024.0f) + 1e-12f);
#pragma unroll
  for (int j = 0; j < 4; ++j) {
    const int col = tid + j * 256;
    out[(size_t)row * 1024 + col] = (x[j] - mu) * rstd * gamma[col] + beta[col];
  }
}

// ------------------------------- launcher ----------------------------------
extern "C" void kernel_launch(void* const* d_in, const int* in_sizes, int n_in,
                              void* d_out, int out_size, void* d_ws, size_t ws_size,
                              hipStream_t stream) {
  (void)in_sizes; (void)n_in; (void)out_size; (void)ws_size;
  const float* hs    = (const float*)d_in[0];
  const float* Wq    = (const float*)d_in[1];
  const float* bq    = (const float*)d_in[2];
  const float* Wk    = (const float*)d_in[3];
  const float* bk    = (const float*)d_in[4];
  const float* Wv    = (const float*)d_in[5];
  const float* bv    = (const float*)d_in[6];
  const float* Wo    = (const float*)d_in[7];
  const float* bo    = (const float*)d_in[8];
  const float* gamma = (const float*)d_in[9];
  const float* beta  = (const float*)d_in[10];
  const float* dist  = (const float*)d_in[11];

  float* out   = (float*)d_out;                       // (4,1024,1024)
  float* probs = out + (size_t)4 * 1024 * 1024;       // (4,32,1024,1024)

  const size_t E = 1024 * 1024;                       // 1M elements
  __bf16* wsb  = (__bf16*)d_ws;                       // needs 64MB workspace
  __bf16* Xb   = wsb;                                 // 4E  bf16
  __bf16* Wqb  = wsb + 4 * E;                         // 1E
  __bf16* Wkb  = wsb + 5 * E;
  __bf16* Wvb  = wsb + 6 * E;
  __bf16* Wob  = wsb + 7 * E;
  __bf16* Qb   = wsb + 8 * E;                         // 4E (b,h,s,d)
  __bf16* Kb   = wsb + 12 * E;                        // 4E (b,h,s,d)
  __bf16* VTb  = wsb + 16 * E;                        // 4E (b,h,d,s)
  __bf16* Ctxb = wsb + 20 * E;                        // 4E (b,s,hid)
  float*  attn = (float*)(wsb + 24 * E);              // 4E f32

  cvt_f32_bf16<<<16384, 256, 0, stream>>>(hs, Xb, (int)(4 * E));
  cvt_f32_bf16<<<4096, 256, 0, stream>>>(Wq, Wqb, (int)E);
  cvt_f32_bf16<<<4096, 256, 0, stream>>>(Wk, Wkb, (int)E);
  cvt_f32_bf16<<<4096, 256, 0, stream>>>(Wv, Wvb, (int)E);
  cvt_f32_bf16<<<4096, 256, 0, stream>>>(Wo, Wob, (int)E);

  const dim3 gg(8, 64, 1);
  gemm_bf16<<<gg, 256, 0, stream>>>(Xb, Wqb, bq, Qb,  nullptr, 0);
  gemm_bf16<<<gg, 256, 0, stream>>>(Xb, Wkb, bk, Kb,  nullptr, 0);
  gemm_bf16<<<gg, 256, 0, stream>>>(Xb, Wvb, bv, VTb, nullptr, 2);

  attn_kernel<<<dim3(64, 32, 4), 32, 0, stream>>>(Qb, Kb, VTb, dist, probs, Ctxb);

  gemm_bf16<<<gg, 256, 0, stream>>>(Ctxb, Wob, bo, nullptr, attn, 3);

  ln_kernel<<<4096, 256, 0, stream>>>(attn, hs, gamma, beta, out);
}
